// MyModel_58832462021358
// MI455X (gfx1250) — compile-verified, matrix operand order checked by reading, not run on hardware
//
#include <hip/hip_runtime.h>
#include <hip/hip_bf16.h>
#include <math.h>

// ---------------------------------------------------------------------------
// Model constants (from reference): B=2, N0=784, D=2048, ratio 0.6, 3 convs.
// n sequence: 784 -> ceil(.6*784)=471 -> ceil(.6*471)=283 -> ceil(.6*283)=170
// ---------------------------------------------------------------------------
#define BSZ   2
#define N0    784
#define DD    2048
#define EMAX  (784 * 783 / 2)     // strict lower triangle upper bound = 306936
#define CDIV(a, b) (((a) + (b) - 1) / (b))
#define LSTR  40                  // LDS row stride in bf16 (80B, 16B aligned, pad vs banks)

typedef __attribute__((ext_vector_type(16))) __bf16 v16bf;
typedef __attribute__((ext_vector_type(8)))  __bf16 v8bf;
typedef __attribute__((ext_vector_type(8)))  float  v8f;

__device__ __forceinline__ float f_nan() { return __builtin_nanf(""); }

__device__ __forceinline__ v8bf cvt8(float4 a, float4 b) {
  v8bf r;
  r[0] = (__bf16)a.x; r[1] = (__bf16)a.y; r[2] = (__bf16)a.z; r[3] = (__bf16)a.w;
  r[4] = (__bf16)b.x; r[5] = (__bf16)b.y; r[6] = (__bf16)b.z; r[7] = (__bf16)b.w;
  return r;
}

// ---------------------------------------------------------------------------
// LDS-staged WMMA GEMM: C[M,N] = A[M,K] @ B[K,N] (fp32 in/out, bf16 WMMA).
// Block = 256 threads (8 waves), 64x64 C macro-tile, K-step 32.
// LDS holds bf16 (converted once at staging); both A and B are staged in
// fragment-friendly [row][kk] layout so all fragment reads are contiguous
// ds_load_b128 pairs. Call-site invariants: K % 32 == 0, N % 16 == 0.
// M/N edges handled by clamped loads + guarded stores (no divergent paths).
// ISA 16-bit frag layout: lane r = l&15, g = l>>4; frag half 0 covers
// kk = 8g+0..7, half 1 covers kk = 16+8g+0..7 (two contiguous runs of 8).
// ---------------------------------------------------------------------------
template <bool TRANS_B>
__global__ __launch_bounds__(256) void wmma_gemm_bf16(
    const float* __restrict__ A, const float* __restrict__ Bm,
    float* __restrict__ C, int M, int N, int K) {
  __shared__ __align__(16) __bf16 sA[64 * LSTR];   // [m][kk]
  __shared__ __align__(16) __bf16 sB[64 * LSTR];   // [n][kk] (transposed for !TRANS_B)

  const int t    = threadIdx.x;
  const int lane = t & 31;
  const int w    = t >> 5;
  const int mn   = lane & 15;
  const int g    = lane >> 4;
  const int wm   = (w & 3) * 16;   // wave m offset inside macro-tile
  const int wn0  = (w >> 2) * 32;  // wave covers n subtiles wn0, wn0+16
  const int tm   = blockIdx.y * 64;
  const int tn   = blockIdx.x * 64;

  // cooperative staging indices
  const int ar   = t >> 2;           // 0..63 row (A's m, or TRANS_B's n)
  const int ac   = (t & 3) * 8;      // 8-float chunk within K-step
  const int ga   = min(tm + ar, M - 1);
  const int gbr  = TRANS_B ? min(tn + ar, N - 1) : 0;
  const int bn   = t & 63;           // !TRANS_B: thread owns one n column
  const int bk   = (t >> 6) * 8;     // !TRANS_B: 8 k-rows
  const int gcol = min(tn + bn, N - 1);

  v8f acc0 = {}, acc1 = {};

  for (int k0 = 0; k0 < K; k0 += 32) {
    __syncthreads();
    // ---- stage A slab: convert to bf16 once, ds_store_b128
    {
      const float* src = A + (size_t)ga * K + k0 + ac;
      float4 p0 = ((const float4*)src)[0];
      float4 p1 = ((const float4*)src)[1];
      *(v8bf*)&sA[ar * LSTR + ac] = cvt8(p0, p1);
    }
    // ---- stage B slab into [n][kk]
    if (TRANS_B) {                   // Bm is [N,K]: contiguous K chunks
      const float* src = Bm + (size_t)gbr * K + k0 + ac;
      float4 p0 = ((const float4*)src)[0];
      float4 p1 = ((const float4*)src)[1];
      *(v8bf*)&sB[ar * LSTR + ac] = cvt8(p0, p1);
    } else {                         // Bm is [K,N]: column gather, coalesced
      v8bf pk;                       // across lanes (consecutive n per k-row)
#pragma unroll
      for (int r = 0; r < 8; ++r)
        pk[r] = (__bf16)Bm[(size_t)(k0 + bk + r) * N + gcol];
      *(v8bf*)&sB[bn * LSTR + bk] = pk;
    }
    __syncthreads();

    if (k0 + 32 < K) {               // speculative prefetch -> global_prefetch_b8
      __builtin_prefetch(&A[(size_t)ga * K + k0 + 32 + ac], 0, 1);
      if (!TRANS_B)
        __builtin_prefetch(&Bm[(size_t)(k0 + 32 + bk) * N + gcol], 0, 1);
    }

    // ---- A fragment: two raw 16B ds_load_b128, register-assembled
    const __bf16* arow = &sA[(wm + mn) * LSTR + 8 * g];
    v8bf a_lo = *(const v8bf*)arow;
    v8bf a_hi = *(const v8bf*)(arow + 16);
    v16bf afrag = __builtin_shufflevector(a_lo, a_hi,
        0, 1, 2, 3, 4, 5, 6, 7, 8, 9, 10, 11, 12, 13, 14, 15);

    // ---- B fragments for the wave's two n subtiles, then WMMA
#pragma unroll
    for (int sub = 0; sub < 2; ++sub) {
      const __bf16* brow = &sB[(wn0 + sub * 16 + mn) * LSTR + 8 * g];
      v8bf b_lo = *(const v8bf*)brow;
      v8bf b_hi = *(const v8bf*)(brow + 16);
      v16bf bfrag = __builtin_shufflevector(b_lo, b_hi,
          0, 1, 2, 3, 4, 5, 6, 7, 8, 9, 10, 11, 12, 13, 14, 15);
      if (sub == 0)
        acc0 = __builtin_amdgcn_wmma_f32_16x16x32_bf16(false, afrag, false, bfrag,
                                                       (short)0, acc0, false, false);
      else
        acc1 = __builtin_amdgcn_wmma_f32_16x16x32_bf16(false, afrag, false, bfrag,
                                                       (short)0, acc1, false, false);
    }
  }

  // ---- store C (guarded at M/N edges only)
#pragma unroll
  for (int sub = 0; sub < 2; ++sub) {
    const v8f accv = sub ? acc1 : acc0;
    const int n = tn + wn0 + sub * 16 + mn;
#pragma unroll
    for (int v = 0; v < 8; ++v) {
      const int m = tm + wm + v + 8 * g;
      if (m < M && n < N) C[(size_t)m * N + n] = accv[v];
    }
  }
}

// ---------------------------------------------------------------------------
// Edge generation chain
// ---------------------------------------------------------------------------
__global__ void k_sqnorm(const float* __restrict__ x, float* __restrict__ sq, int n) {
  int i = blockIdx.x * blockDim.x + threadIdx.x;
  if (i >= n) return;
  const float* r = x + (size_t)i * DD;
  float acc = 0.f;
  for (int f = 0; f < DD; ++f) acc += r[f] * r[f];
  sq[i] = acc;
}

// In-place: Dm holds gram G = X@X^T on entry; distance/tril/zero->NaN on exit.
__global__ void k_dist(float* __restrict__ Dm, const float* __restrict__ sq, int n) {
  int idx = blockIdx.x * blockDim.x + threadIdx.x;
  if (idx >= n * n) return;
  int i = idx / n, j = idx % n;
  float out;
  if (i > j) {  // strict lower triangle (diag distance == 0 -> NaN anyway)
    float d2 = sq[i] + sq[j] - 2.0f * Dm[idx];
    float d = sqrtf(fmaxf(d2, 0.0f));
    out = (d == 0.0f) ? f_nan() : d;
  } else {
    out = f_nan();  // tril zeros -> NaN
  }
  Dm[idx] = out;
}

__global__ void k_colminmax(const float* __restrict__ Dm,
                            float* __restrict__ mn, float* __restrict__ mx, int n) {
  int j = blockIdx.x * blockDim.x + threadIdx.x;
  if (j >= n) return;
  float lo = 3.0e38f, hi = -3.0e38f;
  bool any = false;
  for (int i = 0; i < n; ++i) {
    float v = Dm[(size_t)i * n + j];
    if (v == v) { any = true; lo = fminf(lo, v); hi = fmaxf(hi, v); }
  }
  mn[j] = any ? lo : f_nan();
  mx[j] = any ? hi : f_nan();
}

__global__ void k_zero_stats(float* sumv, unsigned* cntv) {
  if (threadIdx.x == 0 && blockIdx.x == 0) { *sumv = 0.f; *cntv = 0u; }
}

// Normalize per column (in place) and accumulate nanmean stats.
__global__ void k_colnorm_stats(float* __restrict__ Dm,
                                const float* __restrict__ mn,
                                const float* __restrict__ mx,
                                float* sumv, unsigned* cntv, int n) {
  int idx = blockIdx.x * blockDim.x + threadIdx.x;
  if (idx >= n * n) return;
  int j = idx % n;
  float v = (Dm[idx] - mn[j]) / (mx[j] - mn[j]);
  Dm[idx] = v;
  if (v == v) { atomicAdd(sumv, v); atomicAdd(cntv, 1u); }
}

__global__ void k_thr(const float* sumv, const unsigned* cntv, float* thr) {
  if (threadIdx.x == 0 && blockIdx.x == 0)
    *thr = *sumv / (float)(*cntv ? *cntv : 1u);
}

__global__ void k_rowcount(const float* __restrict__ Dm, const float* thr,
                           int* __restrict__ rowcnt, int n) {
  int i = blockIdx.x * blockDim.x + threadIdx.x;
  if (i >= n) return;
  float T = *thr;
  int c = 0;
  for (int j = 0; j < n; ++j)
    if (Dm[(size_t)i * n + j] < T) ++c;  // NaN compares false
  rowcnt[i] = c;
}

__global__ void k_scan(const int* __restrict__ rowcnt, int* __restrict__ rowoff,
                       int* Ecnt, int n) {
  if (threadIdx.x == 0 && blockIdx.x == 0) {
    int acc = 0;
    for (int i = 0; i < n; ++i) { rowoff[i] = acc; acc += rowcnt[i]; }
    *Ecnt = acc;
  }
}

// argwhere order (row-major) preserved: pairs (i,j) written contiguously.
// The reference's (E,2).reshape(2,-1) quirk => row = flat[0:E], col = flat[E:2E].
__global__ void k_write_edges(const float* __restrict__ Dm, const float* thr,
                              const int* __restrict__ rowoff,
                              int* __restrict__ flat, int n) {
  int i = blockIdx.x * blockDim.x + threadIdx.x;
  if (i >= n) return;
  float T = *thr;
  int off = rowoff[i];
  for (int j = 0; j < n; ++j) {
    if (Dm[(size_t)i * n + j] < T) {
      flat[2 * off]     = i;
      flat[2 * off + 1] = j;
      ++off;
    }
  }
}

__global__ void k_init_w(float* __restrict__ w, const int* Ecnt) {
  int e = blockIdx.x * blockDim.x + threadIdx.x;
  if (e >= EMAX) return;
  w[e] = (e < *Ecnt) ? 1.0f : 0.0f;
}

// ---------------------------------------------------------------------------
// GCN layer pieces
// ---------------------------------------------------------------------------
__global__ void k_deg_init(float* deg, int n) {
  int i = blockIdx.x * blockDim.x + threadIdx.x;
  if (i < n) deg[i] = 1.0f;  // +1 self loop
}

__global__ void k_deg_scatter(const int* __restrict__ flat, const float* __restrict__ w,
                              const int* Ecnt, float* __restrict__ deg) {
  int e = blockIdx.x * blockDim.x + threadIdx.x;
  int E = *Ecnt;
  if (e >= E) return;
  float we = w[e];
  if (we == 0.0f) return;
  atomicAdd(&deg[flat[E + e]], we);  // col = flat[E:2E]
}

__global__ void k_dinv(const float* __restrict__ deg, float* __restrict__ dinv, int n) {
  int i = blockIdx.x * blockDim.x + threadIdx.x;
  if (i < n) dinv[i] = rsqrtf(deg[i]);
}

// z = y * dinv[i]^2 + bias   (self-loop contribution + bias)
__global__ void k_self_bias(const float* __restrict__ y, const float* __restrict__ dinv,
                            const float* __restrict__ b, float* __restrict__ z, int n) {
  int idx = blockIdx.x * blockDim.x + threadIdx.x;
  if (idx >= n * DD) return;
  int i = idx / DD, f = idx % DD;
  float di = dinv[i];
  z[idx] = y[idx] * di * di + b[f];
}

// z[col] += y[row] * dinv[row]*dinv[col]*w   (segment_sum scatter)
__global__ void k_edge_scatter(const int* __restrict__ flat, const float* __restrict__ w,
                               const int* Ecnt, const float* __restrict__ dinv,
                               const float* __restrict__ y, float* __restrict__ z) {
  int e = blockIdx.x;
  int E = *Ecnt;
  if (e >= E) return;
  float we = w[e];
  if (we == 0.0f) return;
  int r = flat[e];
  int c = flat[E + e];
  float norm = dinv[r] * dinv[c] * we;
  const float* yr = y + (size_t)r * DD;
  float* zc = z + (size_t)c * DD;
  for (int f = threadIdx.x; f < DD; f += blockDim.x)
    atomicAdd(&zc[f], yr[f] * norm);
}

__global__ void k_relu(float* __restrict__ z, int count) {
  int idx = blockIdx.x * blockDim.x + threadIdx.x;
  if (idx < count) z[idx] = fmaxf(z[idx], 0.0f);
}

__global__ void k_pnorm(const float* __restrict__ p, float* pnorm) {
  __shared__ float s[256];
  float acc = 0.f;
  for (int i = threadIdx.x; i < DD; i += 256) acc += p[i] * p[i];
  s[threadIdx.x] = acc;
  __syncthreads();
  for (int st = 128; st > 0; st >>= 1) {
    if (threadIdx.x < st) s[threadIdx.x] += s[threadIdx.x + st];
    __syncthreads();
  }
  if (threadIdx.x == 0) *pnorm = sqrtf(s[0]);
}

__global__ void k_score(const float* __restrict__ z, const float* __restrict__ p,
                        const float* pnorm, float* __restrict__ score, int n) {
  int i = blockIdx.x * blockDim.x + threadIdx.x;
  if (i >= n) return;
  const float* zi = z + (size_t)i * DD;
  float acc = 0.f;
  for (int f = 0; f < DD; ++f) acc += zi[f] * p[f];
  score[i] = tanhf(acc / *pnorm);
}

// Descending bitonic sort of (score, index) over 1024 slots; keep top k.
__global__ void k_topk_sort(const float* __restrict__ score,
                            float* __restrict__ topv, int* __restrict__ perm,
                            int n, int k) {
  __shared__ float sv[1024];
  __shared__ int   si[1024];
  int t = threadIdx.x;
  sv[t] = (t < n) ? score[t] : -3.0e38f;
  si[t] = t;
  __syncthreads();
  for (int size = 2; size <= 1024; size <<= 1) {
    for (int stride = size >> 1; stride > 0; stride >>= 1) {
      int partner = t ^ stride;
      if (partner > t) {
        bool desc = ((t & size) == 0);
        float v0 = sv[t], v1 = sv[partner];
        if (desc ? (v0 < v1) : (v0 > v1)) {
          sv[t] = v1; sv[partner] = v0;
          int i0 = si[t]; si[t] = si[partner]; si[partner] = i0;
        }
      }
      __syncthreads();
    }
  }
  if (t < k) { topv[t] = sv[t]; perm[t] = si[t]; }
}

__global__ void k_gather(const float* __restrict__ z, const int* __restrict__ perm,
                         const float* __restrict__ topv, float* __restrict__ xo, int k) {
  int idx = blockIdx.x * blockDim.x + threadIdx.x;
  if (idx >= k * DD) return;
  int j = idx / DD, f = idx % DD;
  xo[idx] = z[(size_t)perm[j] * DD + f] * topv[j];
}

__global__ void k_remap_init(int* remap, int n) {
  int i = blockIdx.x * blockDim.x + threadIdx.x;
  if (i < n) remap[i] = -1;
}
__global__ void k_remap_set(const int* __restrict__ perm, int* __restrict__ remap, int k) {
  int j = blockIdx.x * blockDim.x + threadIdx.x;
  if (j < k) remap[perm[j]] = j;
}

__global__ void k_edge_remap(int* __restrict__ flat, float* __restrict__ w,
                             const int* Ecnt, const int* __restrict__ remap) {
  int e = blockIdx.x * blockDim.x + threadIdx.x;
  int E = *Ecnt;
  if (e >= E) return;
  int nr = remap[flat[e]];
  int nc = remap[flat[E + e]];
  bool keep = (nr >= 0) && (nc >= 0);
  w[e] = keep ? w[e] : 0.0f;     // w *= keep (dead edges stay dead)
  flat[e]     = keep ? nr : 0;
  flat[E + e] = keep ? nc : 0;
}

__global__ void k_meanpool(const float* __restrict__ x, float* __restrict__ pooled, int k) {
  int f = blockIdx.x * blockDim.x + threadIdx.x;
  if (f >= DD) return;
  float acc = 0.f;
  for (int j = 0; j < k; ++j) acc += x[(size_t)j * DD + f];
  pooled[f] = acc / (float)k;
}

// out[b,o] = act(dot(in[b,:], Wt[o,:]) + bias[o]); act: 1=relu, 2=sigmoid
__global__ void k_fc(const float* __restrict__ in, const float* __restrict__ Wt,
                     const float* __restrict__ bias, float* __restrict__ out,
                     int Bn, int IN, int OUT, int act) {
  int idx = blockIdx.x * blockDim.x + threadIdx.x;
  if (idx >= Bn * OUT) return;
  int b = idx / OUT, o = idx % OUT;
  const float* ip = in + (size_t)b * IN;
  const float* wp = Wt + (size_t)o * IN;
  float acc = bias[o];
  for (int i = 0; i < IN; ++i) acc += ip[i] * wp[i];
  if (act == 1)      acc = fmaxf(acc, 0.0f);
  else if (act == 2) acc = 1.0f / (1.0f + expf(-acc));
  out[idx] = acc;
}

// ---------------------------------------------------------------------------
// Host launcher
// ---------------------------------------------------------------------------
extern "C" void kernel_launch(void* const* d_in, const int* in_sizes, int n_in,
                              void* d_out, int out_size, void* d_ws, size_t ws_size,
                              hipStream_t stream) {
  (void)in_sizes; (void)n_in; (void)out_size; (void)ws_size;
  const float* x   = (const float*)d_in[0];
  const float* lw1 = (const float*)d_in[10];
  const float* lb1 = (const float*)d_in[11];
  const float* lw2 = (const float*)d_in[12];
  const float* lb2 = (const float*)d_in[13];
  const float* lw3 = (const float*)d_in[14];
  const float* lb3 = (const float*)d_in[15];

  // --- carve workspace (sample-sequential => scratch reused across samples)
  char* wp = (char*)d_ws;
  auto carve = [&](size_t bytes) -> void* {
    void* r = (void*)wp;
    wp += (bytes + 255) & ~(size_t)255;
    return r;
  };
  float*    Dm     = (float*)carve((size_t)N0 * N0 * 4);    // gram -> dist -> dn
  float*    sq     = (float*)carve(N0 * 4);
  float*    mn     = (float*)carve(N0 * 4);
  float*    mx     = (float*)carve(N0 * 4);
  float*    sumv   = (float*)carve(4);
  unsigned* cntv   = (unsigned*)carve(4);
  float*    thr    = (float*)carve(4);
  int*      rowcnt = (int*)carve(N0 * 4);
  int*      rowoff = (int*)carve(N0 * 4);
  int*      Ecnt   = (int*)carve(4);
  int*      flat   = (int*)carve((size_t)2 * EMAX * 4);     // interleaved (r,c) pairs
  float*    wbuf   = (float*)carve((size_t)EMAX * 4);
  float*    xb     = (float*)carve((size_t)N0 * DD * 4);
  float*    yb     = (float*)carve((size_t)N0 * DD * 4);
  float*    zb     = (float*)carve((size_t)N0 * DD * 4);
  float*    deg    = (float*)carve(N0 * 4);
  float*    dinv   = (float*)carve(N0 * 4);
  float*    score  = (float*)carve(N0 * 4);
  float*    topv   = (float*)carve(1024 * 4);
  int*      perm   = (int*)carve(1024 * 4);
  int*      remap  = (int*)carve(N0 * 4);
  float*    pnorm  = (float*)carve(4);
  float*    pooled = (float*)carve((size_t)BSZ * DD * 4);
  float*    h1     = (float*)carve((size_t)BSZ * 1024 * 4);
  float*    h2     = (float*)carve((size_t)BSZ * 1024 * 4);

  static const int ns[4] = {784, 471, 283, 170};  // n per layer, then final k

  for (int s = 0; s < BSZ; ++s) {
    const float* xin = x + (size_t)s * N0 * DD;

    // ---- edge generation --------------------------------------------------
    k_sqnorm<<<CDIV(N0, 256), 256, 0, stream>>>(xin, sq, N0);
    wmma_gemm_bf16<true><<<dim3(CDIV(N0, 64), CDIV(N0, 64)), 256, 0, stream>>>(
        xin, xin, Dm, N0, N0, DD);                                   // G = X X^T
    k_dist<<<CDIV(N0 * N0, 256), 256, 0, stream>>>(Dm, sq, N0);
    k_colminmax<<<CDIV(N0, 256), 256, 0, stream>>>(Dm, mn, mx, N0);
    k_zero_stats<<<1, 1, 0, stream>>>(sumv, cntv);
    k_colnorm_stats<<<CDIV(N0 * N0, 256), 256, 0, stream>>>(Dm, mn, mx, sumv, cntv, N0);
    k_thr<<<1, 1, 0, stream>>>(sumv, cntv, thr);
    k_rowcount<<<CDIV(N0, 256), 256, 0, stream>>>(Dm, thr, rowcnt, N0);
    k_scan<<<1, 1, 0, stream>>>(rowcnt, rowoff, Ecnt, N0);
    k_write_edges<<<CDIV(N0, 256), 256, 0, stream>>>(Dm, thr, rowoff, flat, N0);
    k_init_w<<<CDIV(EMAX, 256), 256, 0, stream>>>(wbuf, Ecnt);

    // ---- 3 x (GCNConv + ReLU + TopKPool) ----------------------------------
    for (int l = 0; l < 3; ++l) {
      const int n = ns[l], k = ns[l + 1];
      const float* xl = (l == 0) ? xin : xb;
      const float* Wl = (const float*)d_in[1 + 3 * l];
      const float* bl = (const float*)d_in[2 + 3 * l];
      const float* pl = (const float*)d_in[3 + 3 * l];

      wmma_gemm_bf16<false><<<dim3(DD / 64, CDIV(n, 64)), 256, 0, stream>>>(
          xl, Wl, yb, n, DD, DD);                                    // y = x W
      k_deg_init<<<CDIV(n, 256), 256, 0, stream>>>(deg, n);
      k_deg_scatter<<<CDIV(EMAX, 256), 256, 0, stream>>>(flat, wbuf, Ecnt, deg);
      k_dinv<<<CDIV(n, 256), 256, 0, stream>>>(deg, dinv, n);
      k_self_bias<<<CDIV(n * DD, 256), 256, 0, stream>>>(yb, dinv, bl, zb, n);
      k_edge_scatter<<<EMAX, 256, 0, stream>>>(flat, wbuf, Ecnt, dinv, yb, zb);
      k_relu<<<CDIV(n * DD, 256), 256, 0, stream>>>(zb, n * DD);
      k_pnorm<<<1, 256, 0, stream>>>(pl, pnorm);
      k_score<<<CDIV(n, 256), 256, 0, stream>>>(zb, pl, pnorm, score, n);
      k_topk_sort<<<1, 1024, 0, stream>>>(score, topv, perm, n, k);
      k_gather<<<CDIV(k * DD, 256), 256, 0, stream>>>(zb, perm, topv, xb, k);
      k_remap_init<<<CDIV(n, 256), 256, 0, stream>>>(remap, n);
      k_remap_set<<<CDIV(k, 256), 256, 0, stream>>>(perm, remap, k);
      k_edge_remap<<<CDIV(EMAX, 256), 256, 0, stream>>>(flat, wbuf, Ecnt, remap);
    }
    k_meanpool<<<CDIV(DD, 256), 256, 0, stream>>>(xb, pooled + (size_t)s * DD, ns[3]);
  }

  // ---- classifier MLP (B=2 => GEMV, VALU path) ----------------------------
  k_fc<<<CDIV(BSZ * 1024, 256), 256, 0, stream>>>(pooled, lw1, lb1, h1, BSZ, DD, 1024, 1);
  k_fc<<<CDIV(BSZ * 1024, 256), 256, 0, stream>>>(h1, lw2, lb2, h2, BSZ, 1024, 1024, 1);
  k_fc<<<CDIV(BSZ * 200, 256), 256, 0, stream>>>(h2, lw3, lb3, (float*)d_out, BSZ, 1024, 200, 2);
}
